// Decoder_50414326120908
// MI455X (gfx1250) — compile-verified, matrix operand order checked by reading, not run on hardware
//
#include <hip/hip_runtime.h>
#include <hip/hip_bf16.h>
#include <stdint.h>
#include <stddef.h>

// ---------------------------------------------------------------------------
// Types for CDNA5 WMMA (wave32): D(16x16 f32) = A(16x32 bf16) x B(32x16 bf16) + C
// ---------------------------------------------------------------------------
typedef __bf16 bf16_t;
typedef __attribute__((ext_vector_type(16))) __bf16 v16bf;
typedef __attribute__((ext_vector_type(8)))  __bf16 v8bf;
typedef __attribute__((ext_vector_type(8)))  float  v8f;

#define BATCH 32
#define DLAT  64

// ---------------------------------------------------------------------------
// CDNA5 async global->LDS copy (ASYNCcnt-tracked DMA, no VGPR staging).
// Prototype (probed from hipcc diagnostic): parameters are pointers to
// int __attribute__((vector_size(16))) in AS(1) (global) and AS(3) (LDS).
// Guarded: falls back to load+ds_store if the builtins are not declared.
// ---------------------------------------------------------------------------
#if defined(__HIP_DEVICE_COMPILE__) && \
    __has_builtin(__builtin_amdgcn_global_load_async_to_lds_b128) && \
    __has_builtin(__builtin_amdgcn_s_wait_asynccnt)
#define USE_ASYNC_COPY 1
#else
#define USE_ASYNC_COPY 0
#endif

typedef int async_v4i __attribute__((vector_size(16)));

__device__ __forceinline__ void async_copy_b128(const bf16_t* g, bf16_t* l)
{
#if USE_ASYNC_COPY
    __builtin_amdgcn_global_load_async_to_lds_b128(
        (__attribute__((address_space(1))) async_v4i*)(void*)const_cast<bf16_t*>(g),
        (__attribute__((address_space(3))) async_v4i*)(void*)l,
        0, 0);
#else
    *(v8bf*)l = *(const v8bf*)g;
#endif
}

__device__ __forceinline__ void async_wait_all()
{
#if USE_ASYNC_COPY
    __builtin_amdgcn_s_wait_asynccnt(0);
#endif
}

// ---------------------------------------------------------------------------
// x3[b,h,w,o] = sum_c code[b,c] * S[b,c,o,h,w]   (NHWC bf16 out, C=3)
// ---------------------------------------------------------------------------
__global__ void latent_decode_kernel(const float* __restrict__ A,
                                     const float* __restrict__ S,
                                     bf16_t* __restrict__ x3)
{
    int t = blockIdx.x * blockDim.x + threadIdx.x;   // (b, h*w)
    if (t >= BATCH * 64 * 64) return;
    int b  = t >> 12;
    int hw = t & 4095;
    float a0 = 0.f, a1 = 0.f, a2 = 0.f;
    const float* Ab = A + b * DLAT;
    const float* Sb = S + (size_t)b * DLAT * 3 * 4096 + hw;
    for (int c = 0; c < DLAT; ++c) {
        float av = Ab[c];
        const float* Sc = Sb + (size_t)c * 3 * 4096;
        a0 += av * Sc[0];
        a1 += av * Sc[4096];
        a2 += av * Sc[2 * 4096];
    }
    bf16_t* o = x3 + (size_t)t * 3;
    o[0] = (bf16_t)a0; o[1] = (bf16_t)a1; o[2] = (bf16_t)a2;
}

// ---------------------------------------------------------------------------
// Repack OIHW fp32 weights -> [Cout][Kpad] bf16, k = (ky*KW+kx)*Cin + c, zero pad
// ---------------------------------------------------------------------------
__global__ void repack_w_kernel(const float* __restrict__ w, bf16_t* __restrict__ out,
                                int Cout, int Cin, int KH, int KW, int Kpad)
{
    int t = blockIdx.x * blockDim.x + threadIdx.x;
    int total = Cout * Kpad;
    if (t >= total) return;
    int n = t / Kpad;
    int k = t - n * Kpad;
    int Ktot = Cin * KH * KW;
    float v = 0.f;
    if (k < Ktot) {
        int c  = k % Cin;
        int r  = k / Cin;
        int kx = r % KW;
        int ky = r / KW;
        v = w[(((size_t)n * Cin + c) * KH + ky) * KW + kx];
    }
    out[t] = (bf16_t)v;
}

// Block-1 conv1 fold: weff[n, (ky*3+kx)*3+o] = sum_d w1[n, d*3+o, ky, kx]  (K=27 pad 32)
__global__ void fold_w1_kernel(const float* __restrict__ w, bf16_t* __restrict__ out)
{
    int t = blockIdx.x * blockDim.x + threadIdx.x;   // 256*32
    if (t >= 256 * 32) return;
    int n = t >> 5, k = t & 31;
    float v = 0.f;
    if (k < 27) {
        int o  = k % 3;
        int r  = k / 3;
        int kx = r % 3;
        int ky = r / 3;
        for (int d = 0; d < DLAT; ++d)
            v += w[(((size_t)n * 192 + d * 3 + o) * 3 + ky) * 3 + kx];
    }
    out[t] = (bf16_t)v;
}

// Block-1 shortcut fold: weff[n, o] = sum_d wsc[n, d*3+o]  (K=3 pad 32)
__global__ void fold_wsc_kernel(const float* __restrict__ w, bf16_t* __restrict__ out)
{
    int t = blockIdx.x * blockDim.x + threadIdx.x;   // 256*32
    if (t >= 256 * 32) return;
    int n = t >> 5, k = t & 31;
    float v = 0.f;
    if (k < 3)
        for (int d = 0; d < DLAT; ++d)
            v += w[(size_t)n * 192 + d * 3 + k];
    out[t] = (bf16_t)v;
}

// BN -> per-channel scale/shift
__global__ void bn_fold_kernel(const float* __restrict__ g, const float* __restrict__ be,
                               const float* __restrict__ m, const float* __restrict__ v,
                               float* __restrict__ scale, float* __restrict__ shift, int C)
{
    int c = blockIdx.x * blockDim.x + threadIdx.x;
    if (c >= C) return;
    float s = g[c] * rsqrtf(v[c] + 1e-5f);
    scale[c] = s;
    shift[c] = be[c] - m[c] * s;
}

// ---------------------------------------------------------------------------
// Implicit-GEMM conv (NHWC bf16 in/out), fused BN scale/shift (+residual)(+ReLU).
// 256 threads = 8 waves; tile BM=128 x BN=128; wave tile 32x64 (2x4 WMMA frags).
// K staged through double-buffered LDS in chunks of 32 using async global->LDS
// DMA so the next chunk's staging overlaps the current chunk's WMMAs.
// ---------------------------------------------------------------------------
#define BM  128
#define BN_ 128
#define BK  32
#define LDT 40   // LDS row stride (bf16) with padding

__global__ __launch_bounds__(256)
void conv_wmma_kernel(const bf16_t* __restrict__ in, int Hin, int Win, int Cin,
                      const bf16_t* __restrict__ wt, int Cout,
                      int KH, int KW, int stride, int pad, int Kpad,
                      const float* __restrict__ scale, const float* __restrict__ shift,
                      const bf16_t* __restrict__ res, int relu,
                      bf16_t* __restrict__ out, int Hout, int Wout)
{
    __shared__ __align__(16) bf16_t ldsA[2][BM * LDT];
    __shared__ __align__(16) bf16_t ldsB[2][BN_ * LDT];

    const int M      = BATCH * Hout * Wout;
    const int HW     = Hout * Wout;
    const int tid    = threadIdx.x;
    const int lane   = tid & 31;
    const int wid    = tid >> 5;
    const int waveM  = (wid & 3) * 32;
    const int waveN  = (wid >> 2) * 64;
    const int blockM = blockIdx.x * BM;
    const int blockN = blockIdx.y * BN_;
    const int Ktot   = Cin * KH * KW;
    const int nchunks = Kpad / BK;
    const bool vecpath = (Cin % 32) == 0;

    // ---- stage one K-chunk into LDS buffer `buf` ----
    auto stage = [&](int ck, int buf) {
        const int kb = ck * BK;
        bf16_t* bufA = ldsA[buf];
        bf16_t* bufB = ldsB[buf];
        // A tile (activations, im2col on the fly)
        if (vecpath) {
            const int rk = kb / Cin;        // (ky*KW+kx), uniform over chunk
            const int c0 = kb % Cin;
            const int kx = rk % KW;
            const int ky = rk / KW;
            #pragma unroll
            for (int i = 0; i < 2; ++i) {
                int vi  = tid + 256 * i;    // 0..511
                int row = vi >> 2;
                int grp = vi & 3;
                int pixel = blockM + row;
                const bf16_t* src = nullptr;
                if (pixel < M) {
                    int b   = pixel / HW;
                    int rem = pixel - b * HW;
                    int ho  = rem / Wout;
                    int wo  = rem - ho * Wout;
                    int hi  = ho * stride - pad + ky;
                    int wi  = wo * stride - pad + kx;
                    if ((unsigned)hi < (unsigned)Hin && (unsigned)wi < (unsigned)Win)
                        src = in + (((size_t)b * Hin + hi) * Win + wi) * Cin + c0 + grp * 8;
                }
                bf16_t* dst = bufA + row * LDT + grp * 8;
                if (src) {
                    async_copy_b128(src, dst);      // ASYNCcnt DMA straight into LDS
                } else {
                    v8bf z;
                    #pragma unroll
                    for (int q = 0; q < 8; ++q) z[q] = (bf16_t)0.0f;
                    *(v8bf*)dst = z;                // halo / M-tail zero fill
                }
            }
        } else {
            // generic gather (only the tiny folded 3-channel layers, nchunks==1)
            #pragma unroll
            for (int i = 0; i < 16; ++i) {
                int e   = tid + 256 * i;    // 0..4095
                int row = e >> 5;
                int col = e & 31;
                int k   = kb + col;
                bf16_t bv = (bf16_t)0.0f;
                int pixel = blockM + row;
                if (k < Ktot && pixel < M) {
                    int c   = k % Cin;
                    int r   = k / Cin;
                    int kx  = r % KW;
                    int ky  = r / KW;
                    int b   = pixel / HW;
                    int rem = pixel - b * HW;
                    int ho  = rem / Wout;
                    int wo  = rem - ho * Wout;
                    int hi  = ho * stride - pad + ky;
                    int wi  = wo * stride - pad + kx;
                    if ((unsigned)hi < (unsigned)Hin && (unsigned)wi < (unsigned)Win)
                        bv = in[(((size_t)b * Hin + hi) * Win + wi) * Cin + c];
                }
                bufA[row * LDT + col] = bv;
            }
        }
        // B tile (pre-repacked [Cout][Kpad] bf16, always dense & in-bounds)
        #pragma unroll
        for (int i = 0; i < 2; ++i) {
            int vi  = tid + 256 * i;
            int row = vi >> 2;
            int grp = vi & 3;
            const bf16_t* src = wt + (size_t)(blockN + row) * Kpad + kb + grp * 8;
            async_copy_b128(src, bufB + row * LDT + grp * 8);
        }
    };

    v8f acc[2][4];
    #pragma unroll
    for (int i = 0; i < 2; ++i)
        #pragma unroll
        for (int j = 0; j < 4; ++j)
            #pragma unroll
            for (int q = 0; q < 8; ++q)
                acc[i][j][q] = 0.f;

    stage(0, 0);

    for (int ck = 0; ck < nchunks; ++ck) {
        const int cur = ck & 1;
        async_wait_all();        // staged DMA for `cur` complete (wave-local)
        __syncthreads();         // publish LDS to all waves

        if (ck + 1 < nchunks)
            stage(ck + 1, cur ^ 1);   // overlap next chunk's DMA with WMMAs

        const bf16_t* bufA = ldsA[cur];
        const bf16_t* bufB = ldsB[cur];
        const int rsel  = lane & 15;
        const int kbase = (lane >> 4) * 8;
        v16bf afrag[2], bfrag[4];
        #pragma unroll
        for (int i = 0; i < 2; ++i) {
            int row = waveM + i * 16 + rsel;
            v8bf lo = *(const v8bf*)&bufA[row * LDT + kbase];
            v8bf hi = *(const v8bf*)&bufA[row * LDT + 16 + kbase];
            afrag[i] = __builtin_shufflevector(lo, hi, 0,1,2,3,4,5,6,7,8,9,10,11,12,13,14,15);
        }
        #pragma unroll
        for (int j = 0; j < 4; ++j) {
            int row = waveN + j * 16 + rsel;
            v8bf lo = *(const v8bf*)&bufB[row * LDT + kbase];
            v8bf hi = *(const v8bf*)&bufB[row * LDT + 16 + kbase];
            bfrag[j] = __builtin_shufflevector(lo, hi, 0,1,2,3,4,5,6,7,8,9,10,11,12,13,14,15);
        }
        #pragma unroll
        for (int i = 0; i < 2; ++i)
            #pragma unroll
            for (int j = 0; j < 4; ++j)
                acc[i][j] = __builtin_amdgcn_wmma_f32_16x16x32_bf16(
                    false, afrag[i], false, bfrag[j], (short)0, acc[i][j], false, false);

        __syncthreads();         // all waves done reading `cur` before restaging it
    }

    // ---- epilogue: BN scale/shift (+residual)(+ReLU), bf16 NHWC store ----
    const int col    = lane & 15;
    const int rowoff = (lane >> 4) * 8;
    #pragma unroll
    for (int j = 0; j < 4; ++j) {
        int n    = blockN + waveN + j * 16 + col;
        float sc = scale[n];
        float sh = shift[n];
        #pragma unroll
        for (int i = 0; i < 2; ++i) {
            #pragma unroll
            for (int r = 0; r < 8; ++r) {
                int pixel = blockM + waveM + i * 16 + rowoff + r;
                if (pixel < M) {
                    float v = acc[i][j][r] * sc + sh;
                    if (res) v += (float)res[(size_t)pixel * Cout + n];
                    if (relu) v = v > 0.f ? v : 0.f;
                    out[(size_t)pixel * Cout + n] = (bf16_t)v;
                }
            }
        }
    }
}

// ---------------------------------------------------------------------------
// Global average pool (16x16) + FC 512->100, fp32 out. One block per batch.
// ---------------------------------------------------------------------------
__global__ void pool_fc_kernel(const bf16_t* __restrict__ act,
                               const float* __restrict__ fcw,
                               const float* __restrict__ fcb,
                               float* __restrict__ out)
{
    __shared__ float pooled[512];
    int b = blockIdx.x, tid = threadIdx.x;
    for (int c = tid; c < 512; c += 256) {
        float s = 0.f;
        const bf16_t* p = act + (size_t)b * 256 * 512 + c;
        for (int px = 0; px < 256; ++px) s += (float)p[(size_t)px * 512];
        pooled[c] = s * (1.f / 256.f);
    }
    __syncthreads();
    for (int n = tid; n < 100; n += 256) {
        float s = fcb[n];
        const float* wr = fcw + n * 512;
        for (int c = 0; c < 512; ++c) s += pooled[c] * wr[c];
        out[b * 100 + n] = s;
    }
}

// ---------------------------------------------------------------------------
// Host-side orchestration
// ---------------------------------------------------------------------------
extern "C" void kernel_launch(void* const* d_in, const int* in_sizes, int n_in,
                              void* d_out, int out_size, void* d_ws, size_t ws_size,
                              hipStream_t stream)
{
    (void)in_sizes; (void)n_in; (void)out_size; (void)ws_size;

    const float* A = (const float*)d_in[0];
    const float* S = (const float*)d_in[1];

    const bool has_sc[5] = { true, true, true, true, false };
    const float* w1[5];  const float* bn1[5][4];
    const float* w2[5];  const float* bn2[5][4];
    const float* wsc[5]; const float* bnsc[5][4];
    int idx = 2;
    for (int i = 0; i < 5; ++i) {
        w1[i] = (const float*)d_in[idx++];
        for (int j = 0; j < 4; ++j) bn1[i][j] = (const float*)d_in[idx++];
        w2[i] = (const float*)d_in[idx++];
        for (int j = 0; j < 4; ++j) bn2[i][j] = (const float*)d_in[idx++];
        if (has_sc[i]) {
            wsc[i] = (const float*)d_in[idx++];
            for (int j = 0; j < 4; ++j) bnsc[i][j] = (const float*)d_in[idx++];
        } else {
            wsc[i] = nullptr;
        }
    }
    const float* fcw = (const float*)d_in[idx++];
    const float* fcb = (const float*)d_in[idx++];

    // ---- bump allocator over d_ws ----
    char* base = (char*)d_ws;
    size_t off = 0;
    auto alloc = [&](size_t bytes) -> void* {
        void* p = base + off;
        off = (off + bytes + 255) & ~(size_t)255;
        return p;
    };

    const size_t ACT = (size_t)BATCH * 64 * 64 * 256 * sizeof(bf16_t);  // 64 MiB
    bf16_t* x3 = (bf16_t*)alloc((size_t)BATCH * 64 * 64 * 3 * sizeof(bf16_t));
    bf16_t* P0 = (bf16_t*)alloc(ACT);
    bf16_t* P1 = (bf16_t*)alloc(ACT);
    bf16_t* T  = (bf16_t*)alloc(ACT);
    bf16_t* SC = (bf16_t*)alloc(ACT);

    // Repacked weights:    conv        Cout  Cin  K  Kpad
    bf16_t* w_l1c1 = (bf16_t*)alloc((size_t)256 * 32   * 2); // folded, K=27->32
    bf16_t* w_l1c2 = (bf16_t*)alloc((size_t)256 * 2304 * 2);
    bf16_t* w_l1sc = (bf16_t*)alloc((size_t)256 * 32   * 2); // folded 1x1, K=3->32
    bf16_t* w_l2c1 = (bf16_t*)alloc((size_t)256 * 2304 * 2);
    bf16_t* w_l2c2 = (bf16_t*)alloc((size_t)256 * 2304 * 2);
    bf16_t* w_l2sc = (bf16_t*)alloc((size_t)256 * 256  * 2);
    bf16_t* w_l3c1 = (bf16_t*)alloc((size_t)512 * 2304 * 2);
    bf16_t* w_l3c2 = (bf16_t*)alloc((size_t)512 * 4608 * 2);
    bf16_t* w_l3sc = (bf16_t*)alloc((size_t)512 * 256  * 2);
    bf16_t* w_l4c1 = (bf16_t*)alloc((size_t)512 * 4608 * 2);
    bf16_t* w_l4c2 = (bf16_t*)alloc((size_t)512 * 4608 * 2);
    bf16_t* w_l4sc = (bf16_t*)alloc((size_t)512 * 512  * 2);
    bf16_t* w_l5c1 = (bf16_t*)alloc((size_t)512 * 4608 * 2);
    bf16_t* w_l5c2 = (bf16_t*)alloc((size_t)512 * 4608 * 2);

    // BN scale/shift, 14 convs in use order
    float* bscale[14]; float* bshift[14];
    const int bnC[14] = {256,256,256, 256,256,256, 512,512,512, 512,512,512, 512,512};
    for (int i = 0; i < 14; ++i) {
        bscale[i] = (float*)alloc(bnC[i] * sizeof(float));
        bshift[i] = (float*)alloc(bnC[i] * sizeof(float));
    }

    // ---- prep kernels ----
    auto repack = [&](const float* w, bf16_t* o, int Cout, int Cin, int KH, int KW, int Kpad) {
        int total = Cout * Kpad;
        repack_w_kernel<<<(total + 255) / 256, 256, 0, stream>>>(w, o, Cout, Cin, KH, KW, Kpad);
    };
    fold_w1_kernel <<<32, 256, 0, stream>>>(w1[0],  w_l1c1);
    fold_wsc_kernel<<<32, 256, 0, stream>>>(wsc[0], w_l1sc);
    repack(w2[0],  w_l1c2, 256, 256, 3, 3, 2304);
    repack(w1[1],  w_l2c1, 256, 256, 3, 3, 2304);
    repack(w2[1],  w_l2c2, 256, 256, 3, 3, 2304);
    repack(wsc[1], w_l2sc, 256, 256, 1, 1, 256);
    repack(w1[2],  w_l3c1, 512, 256, 3, 3, 2304);
    repack(w2[2],  w_l3c2, 512, 512, 3, 3, 4608);
    repack(wsc[2], w_l3sc, 512, 256, 1, 1, 256);
    repack(w1[3],  w_l4c1, 512, 512, 3, 3, 4608);
    repack(w2[3],  w_l4c2, 512, 512, 3, 3, 4608);
    repack(wsc[3], w_l4sc, 512, 512, 1, 1, 512);
    repack(w1[4],  w_l5c1, 512, 512, 3, 3, 4608);
    repack(w2[4],  w_l5c2, 512, 512, 3, 3, 4608);

    auto bnf = [&](const float* const* p, int which, int C) {
        bn_fold_kernel<<<(C + 255) / 256, 256, 0, stream>>>(p[0], p[1], p[2], p[3],
                                                            bscale[which], bshift[which], C);
    };
    bnf(bn1[0], 0, 256);  bnf(bnsc[0], 1, 256);  bnf(bn2[0], 2, 256);
    bnf(bn1[1], 3, 256);  bnf(bnsc[1], 4, 256);  bnf(bn2[1], 5, 256);
    bnf(bn1[2], 6, 512);  bnf(bnsc[2], 7, 512);  bnf(bn2[2], 8, 512);
    bnf(bn1[3], 9, 512);  bnf(bnsc[3],10, 512);  bnf(bn2[3],11, 512);
    bnf(bn1[4],12, 512);                          bnf(bn2[4],13, 512);

    // ---- latent decode ----
    latent_decode_kernel<<<(BATCH * 64 * 64 + 255) / 256, 256, 0, stream>>>(A, S, x3);

    // ---- network ----
    auto conv = [&](const bf16_t* in, int Hin, int Win, int Cin,
                    const bf16_t* wt, int Cout, int KH, int KW, int stride, int pad, int Kpad,
                    int bni, const bf16_t* res, int relu, bf16_t* outp, int Hout, int Wout) {
        int M = BATCH * Hout * Wout;
        dim3 grid((M + BM - 1) / BM, Cout / BN_);
        conv_wmma_kernel<<<grid, 256, 0, stream>>>(in, Hin, Win, Cin, wt, Cout,
                                                   KH, KW, stride, pad, Kpad,
                                                   bscale[bni], bshift[bni], res, relu,
                                                   outp, Hout, Wout);
    };

    // block1: 192(folded->3) -> 256, stride 1, 64x64
    conv(x3, 64, 64, 3,   w_l1c1, 256, 3, 3, 1, 1, 32,   0, nullptr, 1, T,  64, 64);
    conv(x3, 64, 64, 3,   w_l1sc, 256, 1, 1, 1, 0, 32,   1, nullptr, 0, SC, 64, 64);
    conv(T,  64, 64, 256, w_l1c2, 256, 3, 3, 1, 1, 2304, 2, SC,      1, P0, 64, 64);
    // block2: 256 -> 256, stride 2, -> 32x32
    conv(P0, 64, 64, 256, w_l2c1, 256, 3, 3, 2, 1, 2304, 3, nullptr, 1, T,  32, 32);
    conv(P0, 64, 64, 256, w_l2sc, 256, 1, 1, 2, 0, 256,  4, nullptr, 0, SC, 32, 32);
    conv(T,  32, 32, 256, w_l2c2, 256, 3, 3, 1, 1, 2304, 5, SC,      1, P1, 32, 32);
    // block3: 256 -> 512, stride 1, 32x32
    conv(P1, 32, 32, 256, w_l3c1, 512, 3, 3, 1, 1, 2304, 6, nullptr, 1, T,  32, 32);
    conv(P1, 32, 32, 256, w_l3sc, 512, 1, 1, 1, 0, 256,  7, nullptr, 0, SC, 32, 32);
    conv(T,  32, 32, 512, w_l3c2, 512, 3, 3, 1, 1, 4608, 8, SC,      1, P0, 32, 32);
    // block4: 512 -> 512, stride 2, -> 16x16
    conv(P0, 32, 32, 512, w_l4c1, 512, 3, 3, 2, 1, 4608, 9, nullptr, 1, T,  16, 16);
    conv(P0, 32, 32, 512, w_l4sc, 512, 1, 1, 2, 0, 512, 10, nullptr, 0, SC, 16, 16);
    conv(T,  16, 16, 512, w_l4c2, 512, 3, 3, 1, 1, 4608,11, SC,      1, P1, 16, 16);
    // block5: 512 -> 512, stride 1, identity shortcut
    conv(P1, 16, 16, 512, w_l5c1, 512, 3, 3, 1, 1, 4608,12, nullptr, 1, T,  16, 16);
    conv(T,  16, 16, 512, w_l5c2, 512, 3, 3, 1, 1, 4608,13, P1,      1, P0, 16, 16);

    // head: avgpool + fc
    pool_fc_kernel<<<BATCH, 256, 0, stream>>>(P0, fcw, fcb, (float*)d_out);
}
